// Model_Autoencoder_TimeFit_Recursive_31396210933794
// MI455X (gfx1250) — matrix operand order repaired and checked by reading, not compiled, not run on hardware
//
#include <hip/hip_runtime.h>
#include <math.h>

// ---- problem constants (match reference) ----
#define B_SZ   1024
#define P_SZ   512
#define LATN   32
#define HIDN   32
#define NHN    4
#define HDN    8
#define NDN    64
#define N_ITER 10

// d_out layout (floats): time_profile (B*10) | test_chis (B*10) | geom (B*3) | loss (1) | Z (B*32)
#define TP_OFF   0
#define TC_OFF   (B_SZ*10)
#define GEOM_OFF (2*B_SZ*10)
#define LOSS_OFF (2*B_SZ*10 + 3*B_SZ)
#define Z_OFF    (LOSS_OFF + 1)

typedef _Float16 v8h  __attribute__((ext_vector_type(8)));
typedef _Float16 v16h __attribute__((ext_vector_type(16)));
typedef float    v8f  __attribute__((ext_vector_type(8)));

struct Params {
  const float* pix;   const int*   mask;  const float* max_chis; const float* init_latent;
  const float* wq;    const float* bq;    const float* wk;  const float* bk;
  const float* wv;    const float* bv;    const float* wo;  const float* bo;
  const float* wu1;   const float* bu1;   const float* wu2; const float* bu2;
  const float* wg;    const float* bg;    const float* lng; const float* lnb;
  const float* png;   const float* pnb;   const float* wt1; const float* bt1;
  const float* wt2;   const float* bt2;   const float* wt3; const float* bt3;
  const float* wg1;   const float* bg1;   const float* wg2; const float* bg2;
  const float* wg3;   const float* bg3;
  const float* nvalid;
  float* out;
};

__device__ __forceinline__ v16h cat8(v8h lo, v8h hi) {
  return __builtin_shufflevector(lo, hi, 0,1,2,3,4,5,6,7,8,9,10,11,12,13,14,15);
}
// branch-free leaky_relu: for x>0, x >= 0.01x; for x<0, 0.01x > x  -> mul+max, no VCC
__device__ __forceinline__ float leaky(float x) { return fmaxf(x, 0.01f * x); }

// ------------------------------------------------------------------
// Kernel 0: global n_valid = sum(mask) into d_ws[0]
// ------------------------------------------------------------------
__global__ __launch_bounds__(256) void nvalid_kernel(const int* __restrict__ mask, float* __restrict__ out) {
  int tid = blockIdx.x * blockDim.x + threadIdx.x;
  float s = 0.f;
  for (int i = tid; i < B_SZ * P_SZ; i += gridDim.x * blockDim.x) s += (float)mask[i];
  #pragma unroll
  for (int off = 16; off > 0; off >>= 1) s += __shfl_xor(s, off, 32);
  __shared__ float red[8];
  int lane = threadIdx.x & 31, wv = threadIdx.x >> 5;
  if (lane == 0) red[wv] = s;
  __syncthreads();
  if (threadIdx.x == 0) {
    float t = 0.f;
    #pragma unroll
    for (int w = 0; w < 8; ++w) t += red[w];
    atomicAdd(out, t);
  }
}

// ------------------------------------------------------------------
// Main kernel: one workgroup (256 threads = 8 waves) per batch element.
// Full 10-iteration recursion kept in LDS; layer2 of tp_mlp on WMMA f16.
// ------------------------------------------------------------------
__global__ __launch_bounds__(256) void perceiver_timefit_kernel(Params p) {
  const int b    = blockIdx.x;
  const int tid  = threadIdx.x;
  const int lane = tid & 31;
  const int wave = tid >> 5;

  // ---- LDS state ----
  __shared__ float s_chi[P_SZ], s_time[P_SZ], s_maskf[P_SZ], s_diffs[P_SZ];
  __shared__ float s_wk[96], s_wv[96], s_bk[32], s_bv[32];
  __shared__ float s_png[4], s_pnb[4], s_lng[32], s_lnb[32];
  __shared__ float s_Z[32], s_Zn[32], s_Q[32], s_Zc[64], s_w33[64];
  __shared__ float s_pl[32], s_comb[64], s_u1b[32];
  __shared__ float s_bt2[64], s_wt3[64];
  __shared__ float s_gmax[4], s_att[36], s_red[8 * 36];
  __shared__ float s_gbuf[64], s_gbuf2[64];
  __shared__ __align__(16) _Float16 s_fragB[4 * 2 * 32 * 16];   // wt2 pre-swizzled B fragments
  __shared__ __align__(16) _Float16 s_stag[8 * 16 * 72];        // per-wave A staging (pad 64->72)

  // pull the per-batch pixel slab + wt2 toward the WGP (global_prefetch_b8)
  __builtin_prefetch(&p.pix[(size_t)b * P_SZ * 2 + tid * 16], 0, 3);
  __builtin_prefetch(&p.wt2[tid * 16], 0, 3);

  // ---- load per-batch pixel state + iteration-invariant weights ----
  for (int pp = tid; pp < P_SZ; pp += 256) {
    s_chi[pp]   = p.pix[(b * P_SZ + pp) * 2 + 0];
    s_time[pp]  = p.pix[(b * P_SZ + pp) * 2 + 1];
    s_maskf[pp] = (float)p.mask[b * P_SZ + pp];
    s_diffs[pp] = 0.f;
  }
  if (tid < 96) { s_wk[tid] = p.wk[tid]; s_wv[tid] = p.wv[tid]; }
  if (tid < 32) {
    s_bk[tid]  = p.bk[tid];  s_bv[tid] = p.bv[tid];
    s_lng[tid] = p.lng[tid]; s_lnb[tid] = p.lnb[tid];
    s_Z[tid]   = p.init_latent[tid];
  }
  if (tid < 3) { s_png[tid] = p.png[tid]; s_pnb[tid] = p.pnb[tid]; }
  if (tid < 64) {
    s_w33[tid] = p.wt1[tid * 33 + 32];
    s_bt2[tid] = p.bt2[tid];
    s_wt3[tid] = p.wt3[tid];
  }
  // pre-swizzle wt2 into B-fragment order: frag(nt,ks) lane L holds
  // n = nt*16 + (L&15), k = ks*32 + (L<16?0:16) + i, i=0..15 contiguous.
  for (int x = tid; x < 4096; x += 256) {
    int i   = x & 15;
    int ln  = (x >> 4) & 31;
    int ks  = (x >> 9) & 1;
    int nt  = x >> 10;
    int n   = nt * 16 + (ln & 15);
    int k   = ks * 32 + ((ln < 16) ? 0 : 16) + i;
    s_fragB[x] = (_Float16)p.wt2[n * 64 + k];
  }
  __syncthreads();

  const float bt3v = p.bt3[0];
  float loss_local = 0.f;

  // ================= 10-iteration recursion =================
  for (int it = 0; it < N_ITER; ++it) {
    // ---- Zn = LN(Z); Q = Zn @ wq.T + bq ----
    if (tid < 32) {
      float mn = 0.f;
      #pragma unroll
      for (int k = 0; k < 32; ++k) mn += s_Z[k];
      mn *= (1.f / 32.f);
      float var = 0.f;
      #pragma unroll
      for (int k = 0; k < 32; ++k) { float d = s_Z[k] - mn; var += d * d; }
      var *= (1.f / 32.f);
      float inv = rsqrtf(var + 1e-5f);
      s_Zn[tid] = (s_Z[tid] - mn) * inv * s_lng[tid] + s_lnb[tid];
    }
    __syncthreads();
    if (tid < 32) {
      float q = p.bq[tid];
      #pragma unroll
      for (int k = 0; k < 32; ++k) q += p.wq[tid * 32 + k] * s_Zn[k];
      s_Q[tid] = q;
    }
    __syncthreads();

    // ---- per-pixel LN + K projection + logits (2 pixels / thread) ----
    float xns[2][3];
    float lgs[2][4];
    #pragma unroll
    for (int s = 0; s < 2; ++s) {
      int pp = tid + s * 256;
      float x0 = s_chi[pp], x1 = s_time[pp], x2 = s_diffs[pp];
      float mn = (x0 + x1 + x2) * (1.f / 3.f);
      float d0 = x0 - mn, d1 = x1 - mn, d2 = x2 - mn;
      float var = (d0 * d0 + d1 * d1 + d2 * d2) * (1.f / 3.f);
      float inv = rsqrtf(var + 1e-5f);
      float n0 = d0 * inv * s_png[0] + s_pnb[0];
      float n1 = d1 * inv * s_png[1] + s_pnb[1];
      float n2 = d2 * inv * s_png[2] + s_pnb[2];
      xns[s][0] = n0; xns[s][1] = n1; xns[s][2] = n2;
      float qk[4] = {0.f, 0.f, 0.f, 0.f};
      #pragma unroll
      for (int i = 0; i < 32; ++i) {
        float kv = s_wk[i * 3] * n0 + s_wk[i * 3 + 1] * n1 + s_wk[i * 3 + 2] * n2 + s_bk[i];
        qk[i >> 3] += s_Q[i] * kv;
      }
      const float inv8 = 0.35355339059327373f;
      bool msk = (s_maskf[pp] != 0.f);
      #pragma unroll
      for (int h = 0; h < 4; ++h) lgs[s][h] = msk ? qk[h] * inv8 : -1.0e9f;
    }
    // ---- softmax max over P (wave shfl + 8-wave LDS combine) ----
    float mx[4];
    #pragma unroll
    for (int h = 0; h < 4; ++h) mx[h] = fmaxf(lgs[0][h], lgs[1][h]);
    #pragma unroll
    for (int off = 1; off < 32; off <<= 1)
      #pragma unroll
      for (int h = 0; h < 4; ++h) mx[h] = fmaxf(mx[h], __shfl_xor(mx[h], off, 32));
    if (lane == 0) {
      #pragma unroll
      for (int h = 0; h < 4; ++h) s_red[wave * 36 + h] = mx[h];
    }
    __syncthreads();
    if (tid < 4) {
      float m = s_red[tid];
      #pragma unroll
      for (int w = 1; w < 8; ++w) m = fmaxf(m, s_red[w * 36 + tid]);
      s_gmax[tid] = m;
    }
    __syncthreads();

    // ---- exp weights, V projection, pooled accumulation ----
    float pooled[32];
    #pragma unroll
    for (int i = 0; i < 32; ++i) pooled[i] = 0.f;
    float sums[4] = {0.f, 0.f, 0.f, 0.f};
    #pragma unroll
    for (int s = 0; s < 2; ++s) {
      float we[4];
      #pragma unroll
      for (int h = 0; h < 4; ++h) { we[h] = __expf(lgs[s][h] - s_gmax[h]); sums[h] += we[h]; }
      float n0 = xns[s][0], n1 = xns[s][1], n2 = xns[s][2];
      #pragma unroll
      for (int i = 0; i < 32; ++i) {
        float vv = s_wv[i * 3] * n0 + s_wv[i * 3 + 1] * n1 + s_wv[i * 3 + 2] * n2 + s_bv[i];
        pooled[i] += we[i >> 3] * vv;
      }
    }
    #pragma unroll
    for (int off = 1; off < 32; off <<= 1) {
      #pragma unroll
      for (int i = 0; i < 32; ++i) pooled[i] += __shfl_xor(pooled[i], off, 32);
      #pragma unroll
      for (int h = 0; h < 4; ++h) sums[h] += __shfl_xor(sums[h], off, 32);
    }
    if (lane == 0) {
      #pragma unroll
      for (int i = 0; i < 32; ++i) s_red[wave * 36 + i] = pooled[i];
      #pragma unroll
      for (int h = 0; h < 4; ++h) s_red[wave * 36 + 32 + h] = sums[h];
    }
    __syncthreads();
    if (tid < 36) {
      float t = 0.f;
      #pragma unroll
      for (int w = 0; w < 8; ++w) t += s_red[w * 36 + tid];
      s_att[tid] = t;
    }
    __syncthreads();

    // ---- latent update (32 threads) ----
    if (tid < 32) s_pl[tid] = s_att[tid] / s_att[32 + (tid >> 3)];
    __syncthreads();
    if (tid < 32) {
      float acc = p.bo[tid];
      #pragma unroll
      for (int k = 0; k < 32; ++k) acc += p.wo[tid * 32 + k] * s_pl[k];
      s_comb[tid]      = s_Zn[tid];
      s_comb[32 + tid] = acc;
    }
    __syncthreads();
    float gate_r = 0.f;
    if (tid < 32) {
      float a1 = p.bu1[tid], ag = p.bg[tid];
      #pragma unroll
      for (int k = 0; k < 64; ++k) {
        float c = s_comb[k];
        a1 += p.wu1[tid * 64 + k] * c;
        ag += p.wg[tid * 64 + k] * c;
      }
      s_u1b[tid] = 0.5f * a1 * (1.f + erff(a1 * 0.7071067811865475f)); // exact gelu
      gate_r = 1.f / (1.f + __expf(-ag));
    }
    __syncthreads();
    if (tid < 32) {
      float dz = p.bu2[tid];
      #pragma unroll
      for (int k = 0; k < 32; ++k) dz += p.wu2[tid * 32 + k] * s_u1b[k];
      s_Z[tid] = s_Zn[tid] + gate_r * dz;
    }
    __syncthreads();
    // Zc = wt1[:, :32] @ Z + bt1 (rank trick: Z identical across pixels)
    if (tid < 64) {
      float a = p.bt1[tid];
      #pragma unroll
      for (int k = 0; k < 32; ++k) a += p.wt1[tid * 33 + k] * s_Z[k];
      s_Zc[tid] = a;
    }
    __syncthreads();

    // ---- tp_mlp over 512 pixels: layer2 on WMMA (wave owns 4 tiles of 16 px) ----
    {
      const int m    = lane & 15;
      const int kb   = (lane < 16) ? 0 : 16;
      const int nidx = lane & 15;
      v16h bf[4][2];
      #pragma unroll
      for (int nt = 0; nt < 4; ++nt)
        #pragma unroll
        for (int ks = 0; ks < 2; ++ks) {
          const _Float16* fb = &s_fragB[(((nt * 2 + ks) * 32) + lane) * 16];
          bf[nt][ks] = cat8(*(const v8h*)fb, *(const v8h*)(fb + 8));
        }
      for (int rt = 0; rt < 4; ++rt) {
        const int tile = wave * 4 + rt;
        const int p0   = tile * 16;
        _Float16* stg = &s_stag[wave * 16 * 72];
        // layer1 (collapsed): h1 = leaky(Zc + chi * wt1[:,32]) into f16 staging
        #pragma unroll 4
        for (int x = lane; x < 1024; x += 32) {
          int mm = x >> 6, j = x & 63;
          stg[mm * 72 + j] = (_Float16)leaky(s_Zc[j] + s_chi[p0 + mm] * s_w33[j]);
        }
        __builtin_amdgcn_wave_barrier(); // wave-local LDS ordering fence
        const _Float16* row = &stg[m * 72];
        v16h a0 = cat8(*(const v8h*)(row + kb),      *(const v8h*)(row + 8 + kb));
        v16h a1 = cat8(*(const v8h*)(row + 32 + kb), *(const v8h*)(row + 40 + kb));
        float l3[8] = {0.f, 0.f, 0.f, 0.f, 0.f, 0.f, 0.f, 0.f};
        #pragma unroll
        for (int nt = 0; nt < 4; ++nt) {
          float bn = s_bt2[nt * 16 + nidx];
          v8f c = {bn, bn, bn, bn, bn, bn, bn, bn};
          c = __builtin_amdgcn_wmma_f32_16x16x32_f16(false, a0, false, bf[nt][0], (short)0, c, false, false);
          c = __builtin_amdgcn_wmma_f32_16x16x32_f16(false, a1, false, bf[nt][1], (short)0, c, false, false);
          float w3 = s_wt3[nt * 16 + nidx];
          #pragma unroll
          for (int r = 0; r < 8; ++r) l3[r] += leaky(c[r]) * w3; // layer2 act + layer3 partial
        }
        // layer3: reduce over n (16 lanes per half)
        #pragma unroll
        for (int off = 1; off < 16; off <<= 1)
          #pragma unroll
          for (int r = 0; r < 8; ++r) l3[r] += __shfl_xor(l3[r], off, 32);
        if ((lane & 15) == 0) {
          int rbase = (lane >> 4) * 8;
          #pragma unroll
          for (int r = 0; r < 8; ++r) {
            int pp = p0 + rbase + r;
            float err = (l3[r] + bt3v) - s_time[pp]; // reco - input_times
            s_diffs[pp] = err;
            loss_local += s_maskf[pp] * err * err;
          }
        }
      }
    }
    __syncthreads(); // diffs complete before next iteration's pixel LN
  }

  // ================= final outputs =================
  if (tid < 32) p.out[Z_OFF + b * 32 + tid] = s_Z[tid];

  if (tid < 10) {
    float chi = (float)tid * (1.f / 9.f) * p.max_chis[b];
    p.out[TC_OFF + b * 10 + tid] = chi;
    float h1[64];
    #pragma unroll
    for (int j = 0; j < 64; ++j) h1[j] = leaky(s_Zc[j] + chi * s_w33[j]); // s_Zc = final-Z Zc
    float tpv = bt3v;
    for (int j = 0; j < 64; ++j) {
      float a = s_bt2[j];
      #pragma unroll 8
      for (int k = 0; k < 64; ++k) a += p.wt2[j * 64 + k] * h1[k];
      tpv += s_wt3[j] * leaky(a);
    }
    p.out[TP_OFF + b * 10 + tid] = tpv;
  }

  // geometry head
  if (tid < 64) {
    float a = p.bg1[tid];
    #pragma unroll
    for (int k = 0; k < 32; ++k) a += p.wg1[tid * 32 + k] * s_Z[k];
    s_gbuf[tid] = leaky(a);
  }
  __syncthreads();
  if (tid < 64) {
    float a = p.bg2[tid];
    #pragma unroll 8
    for (int k = 0; k < 64; ++k) a += p.wg2[tid * 64 + k] * s_gbuf[k];
    s_gbuf2[tid] = leaky(a);
  }
  __syncthreads();
  if (tid < 3) {
    float a = p.bg3[tid];
    #pragma unroll 8
    for (int k = 0; k < 64; ++k) a += p.wg3[tid * 64 + k] * s_gbuf2[k];
    p.out[GEOM_OFF + b * 3 + tid] = (tid == 0) ? tanhf(a) : a;
  }

  // loss: block reduce then one atomicAdd (iter weight 0.1 folded, /n_valid)
  #pragma unroll
  for (int off = 16; off > 0; off >>= 1) loss_local += __shfl_xor(loss_local, off, 32);
  if (lane == 0) s_red[wave] = loss_local;
  __syncthreads();
  if (tid == 0) {
    float t = 0.f;
    #pragma unroll
    for (int w = 0; w < 8; ++w) t += s_red[w];
    atomicAdd(&p.out[LOSS_OFF], (0.1f * t) / p.nvalid[0]);
  }
}

// ------------------------------------------------------------------
extern "C" void kernel_launch(void* const* d_in, const int* in_sizes, int n_in,
                              void* d_out, int out_size, void* d_ws, size_t ws_size,
                              hipStream_t stream) {
  (void)in_sizes; (void)n_in; (void)out_size; (void)ws_size;
  Params p;
  p.pix         = (const float*)d_in[0];
  p.mask        = (const int*)  d_in[1];
  p.max_chis    = (const float*)d_in[2];
  p.init_latent = (const float*)d_in[3];
  p.wq  = (const float*)d_in[4];  p.bq  = (const float*)d_in[5];
  p.wk  = (const float*)d_in[6];  p.bk  = (const float*)d_in[7];
  p.wv  = (const float*)d_in[8];  p.bv  = (const float*)d_in[9];
  p.wo  = (const float*)d_in[10]; p.bo  = (const float*)d_in[11];
  p.wu1 = (const float*)d_in[12]; p.bu1 = (const float*)d_in[13];
  p.wu2 = (const float*)d_in[14]; p.bu2 = (const float*)d_in[15];
  p.wg  = (const float*)d_in[16]; p.bg  = (const float*)d_in[17];
  p.lng = (const float*)d_in[18]; p.lnb = (const float*)d_in[19];
  p.png = (const float*)d_in[20]; p.pnb = (const float*)d_in[21];
  p.wt1 = (const float*)d_in[22]; p.bt1 = (const float*)d_in[23];
  p.wt2 = (const float*)d_in[24]; p.bt2 = (const float*)d_in[25];
  p.wt3 = (const float*)d_in[26]; p.bt3 = (const float*)d_in[27];
  p.wg1 = (const float*)d_in[28]; p.bg1 = (const float*)d_in[29];
  p.wg2 = (const float*)d_in[30]; p.bg2 = (const float*)d_in[31];
  p.wg3 = (const float*)d_in[32]; p.bg3 = (const float*)d_in[33];
  p.nvalid = (const float*)d_ws;
  p.out    = (float*)d_out;

  hipMemsetAsync(d_ws, 0, sizeof(float), stream);
  hipMemsetAsync((char*)d_out + LOSS_OFF * sizeof(float), 0, sizeof(float), stream);

  hipLaunchKernelGGL(nvalid_kernel, dim3(256), dim3(256), 0, stream,
                     (const int*)d_in[1], (float*)d_ws);
  hipLaunchKernelGGL(perceiver_timefit_kernel, dim3(B_SZ), dim3(256), 0, stream, p);
}
